// SumformerAdjacencyModel_83751862272263
// MI455X (gfx1250) — matrix operand order repaired and checked by reading, not compiled, not run on hardware
//
#include <hip/hip_runtime.h>

typedef __attribute__((ext_vector_type(16))) _Float16 v16h;
typedef __attribute__((ext_vector_type(8)))  _Float16 v8h;
typedef __attribute__((ext_vector_type(8)))  float    v8f;

#define DD 64

// ---------------------------------------------------------------------------
// Stage 1: per-token MLP chain (f32), fold b1 into `a`, emit f16 activations.
// ---------------------------------------------------------------------------
__global__ __launch_bounds__(64) void tok_embed(
    const float* __restrict__ x,
    const float* __restrict__ Wf,  const float* __restrict__ bf,
    const float* __restrict__ Wp1, const float* __restrict__ bp1,
    const float* __restrict__ Wp2, const float* __restrict__ bp2,
    const float* __restrict__ Wn,  const float* __restrict__ bn,
    const float* __restrict__ W1,  const float* __restrict__ b1,
    _Float16* __restrict__ aH, _Float16* __restrict__ bH) {
  __shared__ float p1s[DD], projs[DD], embs[DD];
  const int tok = blockIdx.x;
  const int d   = threadIdx.x;
  const float* xr = x + (size_t)tok * 16;

  float p1 = xr[13] * Wp1[0*DD + d] + xr[14] * Wp1[1*DD + d] +
             xr[15] * Wp1[2*DD + d] + bp1[d];
  p1s[d] = p1 > 0.f ? p1 : 0.f;
  __syncthreads();

  float pos = bp2[d];
  #pragma unroll 8
  for (int k = 0; k < DD; ++k) pos += p1s[k] * Wp2[k*DD + d];

  float proj = bf[d] + pos;
  #pragma unroll
  for (int t = 0; t < 13; ++t) proj += xr[t] * Wf[t*DD + d];
  projs[d] = proj;
  __syncthreads();

  float emb = bn[d];
  #pragma unroll 8
  for (int k = 0; k < DD; ++k) emb += projs[k] * Wn[k*DD + d];
  embs[d] = emb;
  __syncthreads();

  float aa = b1[d], bb = 0.f;
  #pragma unroll 8
  for (int k = 0; k < DD; ++k) {
    float e = embs[k];
    aa += e * W1[k*DD + d];
    bb += e * W1[(DD + k)*DD + d];
  }
  aH[(size_t)tok*DD + d] = (_Float16)aa;
  bH[(size_t)tok*DD + d] = (_Float16)bb;
}

// ---------------------------------------------------------------------------
// Pre-swizzle W2 (64x32 f32) into the wave32 WMMA B-operand layout (f16).
// ---------------------------------------------------------------------------
__global__ __launch_bounds__(128) void swizzle_w2(const float* __restrict__ W2,
                                                  _Float16* __restrict__ W2s) {
  const int t    = threadIdx.x;       // 0..127 = (kt, nt, lane)
  const int kt   = t >> 6;
  const int nt   = (t >> 5) & 1;
  const int lane = t & 31;
  const int kb   = kt*32 + ((lane >> 4) << 3);
  const int n    = nt*16 + (lane & 15);
  _Float16* o = W2s + t*16;
  #pragma unroll
  for (int e = 0; e < 16; ++e) o[e] = (_Float16)W2[(kb + (e < 8 ? e : e + 8))*32 + n];
}

// Register-halving butterfly step: fold register pair (a,b) across lane mask m.
__device__ __forceinline__ float bfly(float a, float b, int m, bool bit) {
  float keep = bit ? b : a;
  float send = bit ? a : b;
  return keep + __shfl_xor(send, m, 32);
}

// ---------------------------------------------------------------------------
// Stage 2: pairwise core. One wave per (batch, 16-wide j tile, 8 i values).
// ---------------------------------------------------------------------------
__global__ __launch_bounds__(256) void pair_kernel(
    const _Float16* __restrict__ aH, const _Float16* __restrict__ bH,
    const _Float16* __restrict__ W2s,
    const float* __restrict__ b2, const float* __restrict__ W3,
    const float* __restrict__ b3, float* __restrict__ out) {
  const int wave = (blockIdx.x * 256 + threadIdx.x) >> 5;  // 0..8191
  const int lid  = threadIdx.x & 31;
  const int ig   = wave & 63;          // 64 groups of 8 i's
  const int jt   = (wave >> 6) & 31;   // j tile (16 wide)
  const int bat  = wave >> 11;         // batch 0..3
  const int j0   = jt * 16;
  const int row  = lid & 15;           // pair-row inside tile (j = j0+row)
  const int hi   = lid >> 4;           // 0: K{0..7,16..23}, 1: K{8..15,24..31}

  const v8h* bj = (const v8h*)(bH + (size_t)(bat*512 + j0 + row) * DD);
  const v8h bv0 = bj[hi + 0];
  const v8h bv1 = bj[hi + 2];
  const v8h bv2 = bj[hi + 4];
  const v8h bv3 = bj[hi + 6];

  const v16h* w2p = (const v16h*)W2s;
  const v16h B00 = w2p[ 0 + lid];   // K 0..31,  N 0..15
  const v16h B01 = w2p[32 + lid];   // K 0..31,  N 16..31
  const v16h B10 = w2p[64 + lid];   // K 32..63, N 0..15
  const v16h B11 = w2p[96 + lid];   // K 32..63, N 16..31

  const int   n     = lid & 15;
  const float b2v0  = b2[n];
  const float b2v1  = b2[n + 16];
  const float w30   = W3[n];
  const float w31   = W3[n + 16];
  const float bias3 = b3[0];

  const bool bit0 = (lid & 1) != 0;
  const bool bit1 = (lid & 2) != 0;
  const bool bit2 = (lid & 4) != 0;
  const bool writer = (lid & 8) == 0;   // lanes 0-7 / 16-23 store rows
  const int  outcol = (lid & 7) + hi * 8;

  #pragma unroll 2
  for (int ii = 0; ii < 8; ++ii) {
    const int i = ig*8 + ii;
    const v8h* ai = (const v8h*)(aH + (size_t)(bat*512 + i) * DD);
    const v8h av0 = ai[hi + 0];
    const v8h av1 = ai[hi + 2];
    const v8h av2 = ai[hi + 4];
    const v8h av3 = ai[hi + 6];

    // h = relu(a + b) in packed f16 (b1 folded into a)
    v8h h0, h1, h2, h3;
    #pragma unroll
    for (int e = 0; e < 8; ++e) {
      const _Float16 z = (_Float16)0;
      _Float16 s0 = (_Float16)(av0[e] + bv0[e]); h0[e] = s0 > z ? s0 : z;
      _Float16 s1 = (_Float16)(av1[e] + bv1[e]); h1[e] = s1 > z ? s1 : z;
      _Float16 s2 = (_Float16)(av2[e] + bv2[e]); h2[e] = s2 > z ? s2 : z;
      _Float16 s3 = (_Float16)(av3[e] + bv3[e]); h3[e] = s3 > z ? s3 : z;
    }
    v16h A0, A1;   // A-operand per ISA 16-bit 16x32 layout
    #pragma unroll
    for (int e = 0; e < 8; ++e) {
      A0[e] = h0[e]; A0[e + 8] = h1[e];
      A1[e] = h2[e]; A1[e + 8] = h3[e];
    }

    v8f c0 = {}; v8f c1 = {};
    c0 = __builtin_amdgcn_wmma_f32_16x16x32_f16(false, A0, false, B00, (short)0, c0, false, false);
    c0 = __builtin_amdgcn_wmma_f32_16x16x32_f16(false, A1, false, B10, (short)0, c0, false, false);
    c1 = __builtin_amdgcn_wmma_f32_16x16x32_f16(false, A0, false, B01, (short)0, c1, false, false);
    c1 = __builtin_amdgcn_wmma_f32_16x16x32_f16(false, A1, false, B11, (short)0, c1, false, false);

    // t_r = relu(D + b2) @ W3 per row r (VGPR index), then register-halving
    // butterfly: after masks 1,2,4 each lane holds the 8-lane partial for
    // row r = lid&7; mask 8 completes the 16-lane sum.
    float t0_ = c0[0] + b2v0; t0_ = t0_ > 0.f ? t0_ : 0.f;
    float t1_ = c1[0] + b2v1; t1_ = t1_ > 0.f ? t1_ : 0.f;
    float t0 = t0_ * w30 + t1_ * w31;
    float t1, t2, t3, t4, t5, t6, t7;
    { float a_ = c0[1] + b2v0; a_ = a_ > 0.f ? a_ : 0.f;
      float b_ = c1[1] + b2v1; b_ = b_ > 0.f ? b_ : 0.f; t1 = a_*w30 + b_*w31; }
    { float a_ = c0[2] + b2v0; a_ = a_ > 0.f ? a_ : 0.f;
      float b_ = c1[2] + b2v1; b_ = b_ > 0.f ? b_ : 0.f; t2 = a_*w30 + b_*w31; }
    { float a_ = c0[3] + b2v0; a_ = a_ > 0.f ? a_ : 0.f;
      float b_ = c1[3] + b2v1; b_ = b_ > 0.f ? b_ : 0.f; t3 = a_*w30 + b_*w31; }
    { float a_ = c0[4] + b2v0; a_ = a_ > 0.f ? a_ : 0.f;
      float b_ = c1[4] + b2v1; b_ = b_ > 0.f ? b_ : 0.f; t4 = a_*w30 + b_*w31; }
    { float a_ = c0[5] + b2v0; a_ = a_ > 0.f ? a_ : 0.f;
      float b_ = c1[5] + b2v1; b_ = b_ > 0.f ? b_ : 0.f; t5 = a_*w30 + b_*w31; }
    { float a_ = c0[6] + b2v0; a_ = a_ > 0.f ? a_ : 0.f;
      float b_ = c1[6] + b2v1; b_ = b_ > 0.f ? b_ : 0.f; t6 = a_*w30 + b_*w31; }
    { float a_ = c0[7] + b2v0; a_ = a_ > 0.f ? a_ : 0.f;
      float b_ = c1[7] + b2v1; b_ = b_ > 0.f ? b_ : 0.f; t7 = a_*w30 + b_*w31; }

    float u0 = bfly(t0, t1, 1, bit0);
    float u1 = bfly(t2, t3, 1, bit0);
    float u2 = bfly(t4, t5, 1, bit0);
    float u3 = bfly(t6, t7, 1, bit0);
    float v0 = bfly(u0, u1, 2, bit1);
    float v1 = bfly(u2, u3, 2, bit1);
    float w  = bfly(v0, v1, 4, bit2);
    float s  = w + __shfl_xor(w, 8, 32);

    if (writer) {
      const size_t base = ((size_t)(bat*512 + i))*512 + (size_t)(j0 + outcol);
      out[base] = s + bias3;
    }
  }
}

extern "C" void kernel_launch(void* const* d_in, const int* in_sizes, int n_in,
                              void* d_out, int out_size, void* d_ws, size_t ws_size,
                              hipStream_t stream) {
  const float* x   = (const float*)d_in[0];
  const float* Wf  = (const float*)d_in[1];
  const float* bf  = (const float*)d_in[2];
  const float* Wp1 = (const float*)d_in[3];
  const float* bp1 = (const float*)d_in[4];
  const float* Wp2 = (const float*)d_in[5];
  const float* bp2 = (const float*)d_in[6];
  const float* Wn  = (const float*)d_in[7];
  const float* bn  = (const float*)d_in[8];
  const float* W1  = (const float*)d_in[9];
  const float* b1  = (const float*)d_in[10];
  const float* W2  = (const float*)d_in[11];
  const float* b2  = (const float*)d_in[12];
  const float* W3  = (const float*)d_in[13];
  const float* b3  = (const float*)d_in[14];

  _Float16* aH  = (_Float16*)d_ws;          // 2048*64 halfs
  _Float16* bH  = aH + 2048 * DD;           // 2048*64 halfs
  _Float16* W2s = bH + 2048 * DD;           // 2048 halfs (swizzled W2)

  hipLaunchKernelGGL(tok_embed, dim3(2048), dim3(64), 0, stream,
                     x, Wf, bf, Wp1, bp1, Wp2, bp2, Wn, bn, W1, b1, aH, bH);
  hipLaunchKernelGGL(swizzle_w2, dim3(1), dim3(128), 0, stream, W2, W2s);
  hipLaunchKernelGGL(pair_kernel, dim3(1024), dim3(256), 0, stream,
                     aH, bH, W2s, b2, W3, b3, (float*)d_out);
}
